// EducationalMultiHead_40862318854579
// MI455X (gfx1250) — matrix operand order repaired and checked by reading, not compiled, not run on hardware
//
#include <hip/hip_runtime.h>

// ---------------------------------------------------------------------------
// MI455X (gfx1250, wave32) fused multi-head attention:
//   K1: QKV GEMM (f32 -> f16 WMMA; V written pre-transposed [b,h,d,s])
//   K2: flash attention (TDM tensor_load_to_lds staging, WMMA QK^T,
//       ds_swizzle row-max, WMMA row-sums via ones-matrix, WMMA PV)
//   K3: output projection GEMM (+bias, f32 out)
// ---------------------------------------------------------------------------

typedef __attribute__((ext_vector_type(16))) _Float16 v16h;
typedef __attribute__((ext_vector_type(8)))  _Float16 v8h;
typedef __attribute__((ext_vector_type(4)))  _Float16 v4h;
typedef __attribute__((ext_vector_type(8)))  float    v8f;
typedef __attribute__((ext_vector_type(4)))  int      v4i;

#define WMMA_F16(A, B, C) \
  __builtin_amdgcn_wmma_f32_16x16x32_f16(false, (A), false, (B), (short)0, (C), false, false)

#if __has_builtin(__builtin_amdgcn_global_load_async_to_lds_b128) && \
    __has_builtin(__builtin_amdgcn_s_wait_asynccnt)
#define USE_ASYNC_LDS 1
typedef __attribute__((address_space(1))) v4i* g_v4i;  // global v4i*
typedef __attribute__((address_space(3))) v4i* l_v4i;  // LDS v4i*
#define ASYNC_LDS_B128(gsrc, ldst) \
  __builtin_amdgcn_global_load_async_to_lds_b128((g_v4i)(void*)(gsrc), \
                                                 (l_v4i)(void*)(ldst), 0, 0)
#else
#define USE_ASYNC_LDS 0
#endif

#if __has_builtin(__builtin_amdgcn_tensor_load_to_lds) && \
    __has_builtin(__builtin_amdgcn_s_wait_tensorcnt)
#define USE_TDM 1
typedef __attribute__((ext_vector_type(4))) unsigned int v4u;
typedef __attribute__((ext_vector_type(8))) int v8i;

__device__ __forceinline__ unsigned lds_off_u32(const void* p) {
  return (unsigned)(size_t)(__attribute__((address_space(3))) const void*)p;
}

// 2D TDM copy, f16 elements, +16B LDS padding per tile row.
// D# per CDNA5 ISA ch.8: group0 = {count=1, lds_addr, global_addr, type=2};
// group1 = {data_size=1(2B), pad_enable, pad_interval code, pad_amount=4DW,
//           tensor_dim0/1, tile_dim0/1, tensor_dim0_stride}; groups 2/3 = 0.
__device__ __forceinline__ void tdm_load_2d(unsigned lds, unsigned long long ga,
                                            unsigned pad_interval_code,
                                            unsigned td0, unsigned td1,
                                            unsigned tile0, unsigned tile1,
                                            unsigned stride0) {
  v4u g0 = {1u,                                       // count=1, user mode
            lds,                                      // lds_addr[63:32]
            (unsigned)(ga & 0xffffffffu),             // global_addr lo
            (unsigned)((ga >> 32) & 0x01ffffffu) | (2u << 30)};  // type=2
  unsigned f = (1u << 16)                 // data_size = 1 -> 2 bytes
             | (1u << 20)                 // pad_enable
             | (pad_interval_code << 22)  // DWORDs between pads
             | (3u << 25);                // pad_amount code 3 -> 4 DWORDs (16B)
  v8i g1 = {(int)f,
            (int)((td0 & 0xffffu) << 16),                               // [63:48]
            (int)(((td0 >> 16) & 0xffffu) | ((td1 & 0xffffu) << 16)),   // [95:64]
            (int)(((td1 >> 16) & 0xffffu) | ((tile0 & 0xffffu) << 16)), // [127:96]
            (int)(tile1 & 0xffffu),                                     // tile_dim1/2
            (int)stride0,                                               // stride0 lo
            0, 0};
  v4i gz = {0, 0, 0, 0};
  v8i gz8 = {0, 0, 0, 0, 0, 0, 0, 0};
  __builtin_amdgcn_tensor_load_to_lds(g0, g1, gz, gz, gz8, 0);
}
#else
#define USE_TDM 0
#endif

// butterfly max across the 16 lanes of a half-wave (xor 1,2,4,8 stay in-half)
__device__ __forceinline__ float halfrow_max(float v) {
#if __has_builtin(__builtin_amdgcn_ds_swizzle)
  v = fmaxf(v, __int_as_float(__builtin_amdgcn_ds_swizzle(__float_as_int(v), 0x041F)));
  v = fmaxf(v, __int_as_float(__builtin_amdgcn_ds_swizzle(__float_as_int(v), 0x081F)));
  v = fmaxf(v, __int_as_float(__builtin_amdgcn_ds_swizzle(__float_as_int(v), 0x101F)));
  v = fmaxf(v, __int_as_float(__builtin_amdgcn_ds_swizzle(__float_as_int(v), 0x201F)));
#else
  v = fmaxf(v, __shfl_xor(v, 1));
  v = fmaxf(v, __shfl_xor(v, 2));
  v = fmaxf(v, __shfl_xor(v, 4));
  v = fmaxf(v, __shfl_xor(v, 8));
#endif
  return v;
}

// A fragment (16x32 f16, MxK): lane l -> row m=l%16; halves 0-7 -> k=base8+0..7,
// halves 8-15 -> k=base8+16..23, base8=(l>=16)?8:0. `row` = 16B-aligned row start.
__device__ __forceinline__ v16h frag_a_k32(const _Float16* row, int lane) {
  const int off = (lane >= 16) ? 8 : 0;
  v8h lo = *(const v8h*)(row + off);
  v8h hi = *(const v8h*)(row + off + 16);
  return __builtin_shufflevector(lo, hi, 0,1,2,3,4,5,6,7,8,9,10,11,12,13,14,15);
}

// B fragment (32x16 f16, KxN), K-contiguous per column: lane l -> col n=l%16;
// halves 0-7 -> k=base8+0..7, halves 8-15 -> k=base8+16..23, base8=(l>=16)?8:0.
__device__ __forceinline__ v16h frag_b_k32(const _Float16* row, int lane) {
  const int off = (lane >= 16) ? 8 : 0;
  v8h lo = *(const v8h*)(row + off);
  v8h hi = *(const v8h*)(row + off + 16);
  return __builtin_shufflevector(lo, hi, 0,1,2,3,4,5,6,7,8,9,10,11,12,13,14,15);
}

// ---------------------------------------------------------------------------
// 128x128 block-tile GEMM, BK=32, 256 threads (8 wave32 waves), 2x4 wave grid.
// MODE 0: QKV epilogue -> f16 Q[b,h,s,d], K[b,h,s,d], V transposed [b,h,d,s]
// MODE 1: proj epilogue -> f32 out [M,N] + bias
// ---------------------------------------------------------------------------
template <int MODE>
__global__ __launch_bounds__(256) void hgemm_wmma_gfx1250(
    const float* __restrict__ A, const float* __restrict__ W,
    const float* __restrict__ bias,
    _Float16* __restrict__ qb, _Float16* __restrict__ kbuf,
    _Float16* __restrict__ vbuf, float* __restrict__ out,
    int M, int N, int K) {
  __shared__ __attribute__((aligned(16))) _Float16 As[128][40];   // [m][k]
  __shared__ __attribute__((aligned(16))) _Float16 BsT[128][40];  // [n][k]

  const int t = threadIdx.x;
  const int lane = t & 31;
  const int wave = t >> 5;
  const int wm = (wave & 1) * 64;
  const int wn = (wave >> 1) * 32;
  const int bm = blockIdx.y * 128;
  const int bn = blockIdx.x * 128;

  const v8f vz = {0.f, 0.f, 0.f, 0.f, 0.f, 0.f, 0.f, 0.f};
  v8f acc[4][2];
#pragma unroll
  for (int mi = 0; mi < 4; ++mi)
#pragma unroll
    for (int ni = 0; ni < 2; ++ni) acc[mi][ni] = vz;

  for (int k0 = 0; k0 < K; k0 += 32) {
    // ---- stage A tile (128x32 f32 -> f16), coalesced float4 loads ----
#pragma unroll
    for (int i = 0; i < 4; ++i) {
      int flat = t + i * 256;
      int m = flat >> 3;
      int kg = (flat & 7) * 4;
      float4 av = *(const float4*)(A + (size_t)(bm + m) * K + k0 + kg);
      v4h hv = {(_Float16)av.x, (_Float16)av.y, (_Float16)av.z, (_Float16)av.w};
      *(v4h*)&As[m][kg] = hv;
    }
    // ---- stage W tile transposed (32x128 -> BsT[n][k]) ----
#pragma unroll
    for (int i = 0; i < 4; ++i) {
      int flat = t + i * 256;
      int kg = (flat >> 7) * 4;
      int n = flat & 127;
      const float* wp = W + (size_t)(k0 + kg) * N + bn + n;
      v4h hv = {(_Float16)wp[0], (_Float16)wp[(size_t)N],
                (_Float16)wp[2 * (size_t)N], (_Float16)wp[3 * (size_t)N]};
      *(v4h*)&BsT[n][kg] = hv;
    }
    if (k0 + 32 < K) {  // global_prefetch_b8 for the next K tile
      __builtin_prefetch(A + (size_t)(bm + (t >> 3)) * K + k0 + 32, 0, 0);
      __builtin_prefetch(W + (size_t)(k0 + 32 + (t >> 7)) * N + bn + (t & 127), 0, 0);
    }
    __syncthreads();

    v16h bf[2];
#pragma unroll
    for (int ni = 0; ni < 2; ++ni)
      bf[ni] = frag_b_k32(&BsT[wn + ni * 16 + (lane & 15)][0], lane);
#pragma unroll
    for (int mi = 0; mi < 4; ++mi) {
      v16h af = frag_a_k32(&As[wm + mi * 16 + (lane & 15)][0], lane);
#pragma unroll
      for (int ni = 0; ni < 2; ++ni)
        acc[mi][ni] = WMMA_F16(af, bf[ni], acc[mi][ni]);
    }
    __syncthreads();
  }

  // ---- epilogue (C layout: VGPR r, lanes0-15 -> M=r, lanes16-31 -> M=8+r) --
  const int halfoff = (lane >= 16) ? 8 : 0;
#pragma unroll
  for (int mi = 0; mi < 4; ++mi)
#pragma unroll
    for (int ni = 0; ni < 2; ++ni)
#pragma unroll
      for (int r = 0; r < 8; ++r) {
        int gm = bm + wm + mi * 16 + r + halfoff;
        int gn = bn + wn + ni * 16 + (lane & 15);
        float val = acc[mi][ni][r] + bias[gn];
        if (MODE == 0) {
          int bi = gm >> 11, s = gm & 2047;    // row -> (batch, seq)
          int ty = gn >> 10, rem = gn & 1023;  // col -> (qkv, head, d)
          int h = rem >> 6, d = rem & 63;
          size_t hb = (size_t)bi * 16 + h;
          if (ty == 0)
            qb[(hb * 2048 + s) * 64 + d] = (_Float16)val;
          else if (ty == 1)
            kbuf[(hb * 2048 + s) * 64 + d] = (_Float16)val;
          else  // V stored transposed: [b,h,d,s]
            vbuf[(hb * 64 + d) * 2048 + s] = (_Float16)val;
        } else {
          out[(size_t)gm * N + gn] = val;
        }
      }
}

// ---------------------------------------------------------------------------
// Flash attention: block = 64 queries of one (batch,head); 4 waves, 16 q/wave.
// Per 32-key block: TDM-stage K[32][64] and V^T[64][32] into LDS (one
// tensor_load_to_lds per tile, issued by thread 0, s_wait_tensorcnt), 4 WMMA
// for S, online softmax (ds_swizzle row-max; row-sum via ones-matrix WMMA),
// 5 WMMA for [O|rowsum] += P.[V|1].
// ---------------------------------------------------------------------------
__global__ __launch_bounds__(128) void flash_attn_wmma_gfx1250(
    const _Float16* __restrict__ qb, const _Float16* __restrict__ kb,
    const _Float16* __restrict__ vb, float* __restrict__ attn) {
  __shared__ __attribute__((aligned(16))) _Float16 Ks[32][72];    // [key][d]
  __shared__ __attribute__((aligned(16))) _Float16 VsT[64][40];   // [d][key]
  __shared__ __attribute__((aligned(16))) _Float16 Ps[4][16][40]; // per-wave P

  const int t = threadIdx.x;
  const int lane = t & 31;
  const int w = t >> 5;
  const int bh = blockIdx.y;  // b*16 + h
  const int q0 = blockIdx.x * 64;
  const size_t headbase = (size_t)bh * 2048 * 64;
  const int halfoff = (lane >= 16) ? 8 : 0;

  // Q fragments, scale 1/sqrt(64)=0.125 folded in (exact in f16)
  const _Float16* qrow = qb + headbase + (size_t)(q0 + w * 16 + (lane & 15)) * 64;
  v16h qa0 = frag_a_k32(qrow, lane);
  v16h qa1 = frag_a_k32(qrow + 32, lane);
  v16h ones;
#pragma unroll
  for (int i = 0; i < 16; ++i) {
    qa0[i] = qa0[i] * (_Float16)0.125f;
    qa1[i] = qa1[i] * (_Float16)0.125f;
    ones[i] = (_Float16)1.0f;
  }

  const v8f vz = {0.f, 0.f, 0.f, 0.f, 0.f, 0.f, 0.f, 0.f};
  v8f o[5];  // o[0..3] = output d-chunks, o[4] = row sums (ones-matrix WMMA)
#pragma unroll
  for (int j = 0; j < 5; ++j) o[j] = vz;
  float rmax[8];
#pragma unroll
  for (int r = 0; r < 8; ++r) rmax[r] = -1.0e30f;

#if USE_TDM
  const unsigned ks_lds = lds_off_u32(&Ks[0][0]);
  const unsigned vs_lds = lds_off_u32(&VsT[0][0]);
#endif

  for (int k0 = 0; k0 < 2048; k0 += 32) {
    // ---- stage K block [32][64] and V^T block [64][32] ----
#if USE_TDM
    if (t == 0) {
      // K: 32 rows x 64 f16 (128B data + 16B pad per row -> Ks stride 72h)
      tdm_load_2d(ks_lds,
                  (unsigned long long)(size_t)(kb + headbase + (size_t)k0 * 64),
                  /*pad_interval 32 DW*/ 4, /*td0*/ 64, /*td1*/ 2048,
                  /*tile0*/ 64, /*tile1*/ 32, /*stride0*/ 64);
      // V^T: 64 rows x 32 f16 (64B data + 16B pad per row -> VsT stride 40h)
      tdm_load_2d(vs_lds,
                  (unsigned long long)(size_t)(vb + headbase + (size_t)k0),
                  /*pad_interval 16 DW*/ 3, /*td0*/ 2048, /*td1*/ 64,
                  /*tile0*/ 32, /*tile1*/ 64, /*stride0*/ 2048);
      __builtin_amdgcn_s_wait_tensorcnt(0);
    }
#elif USE_ASYNC_LDS
#pragma unroll
    for (int i = 0; i < 2; ++i) {  // 256 x 16B chunks of K, 2 per thread
      int c = t + i * 128;
      int key = c >> 3, d = (c & 7) * 8;
      ASYNC_LDS_B128(kb + headbase + (size_t)(k0 + key) * 64 + d, &Ks[key][d]);
    }
#pragma unroll
    for (int i = 0; i < 2; ++i) {  // 256 x 16B chunks of V^T, 2 per thread
      int c = t + i * 128;
      int d = c >> 2, key = (c & 3) * 8;
      ASYNC_LDS_B128(vb + headbase + (size_t)d * 2048 + k0 + key, &VsT[d][key]);
    }
    __builtin_amdgcn_s_wait_asynccnt(0);
#else
#pragma unroll
    for (int i = 0; i < 2; ++i) {
      int c = t + i * 128;
      int key = c >> 3, d = (c & 7) * 8;
      *(v8h*)&Ks[key][d] =
          *(const v8h*)(kb + headbase + (size_t)(k0 + key) * 64 + d);
    }
#pragma unroll
    for (int i = 0; i < 2; ++i) {
      int c = t + i * 128;
      int d = c >> 2, key = (c & 3) * 8;
      *(v8h*)&VsT[d][key] =
          *(const v8h*)(vb + headbase + (size_t)d * 2048 + k0 + key);
    }
#endif
    __syncthreads();

    // ---- S tiles: 16q x 32keys as two 16x16 C frags ----
    v8f s0 = vz, s1 = vz;
    {
      const int kr = lane & 15;
      v16h b0 = frag_b_k32(&Ks[kr][0], lane);
      v16h b1 = frag_b_k32(&Ks[kr][32], lane);
      s0 = WMMA_F16(qa0, b0, s0);
      s0 = WMMA_F16(qa1, b1, s0);
      v16h c0 = frag_b_k32(&Ks[16 + kr][0], lane);
      v16h c1 = frag_b_k32(&Ks[16 + kr][32], lane);
      s1 = WMMA_F16(qa0, c0, s1);
      s1 = WMMA_F16(qa1, c1, s1);
    }

    // ---- online softmax (rows live in 16-lane halves) ----
#pragma unroll
    for (int r = 0; r < 8; ++r) {
      float mx = halfrow_max(fmaxf(s0[r], s1[r]));
      float nm = fmaxf(rmax[r], mx);
      float corr = __expf(rmax[r] - nm);
      rmax[r] = nm;
      float p0 = __expf(s0[r] - nm);
      float p1 = __expf(s1[r] - nm);
#pragma unroll
      for (int j = 0; j < 5; ++j) o[j][r] *= corr;
      int pr = r + halfoff;  // C layout -> wave-private LDS tile
      Ps[w][pr][lane & 15] = (_Float16)p0;
      Ps[w][pr][16 + (lane & 15)] = (_Float16)p1;
    }

    // ---- [O | rowsum] += P . [V | ones] ----
    v16h pa = frag_a_k32(&Ps[w][lane & 15][0], lane);
#pragma unroll
    for (int j = 0; j < 4; ++j) {
      v16h vf = frag_b_k32(&VsT[j * 16 + (lane & 15)][0], lane);
      o[j] = WMMA_F16(pa, vf, o[j]);
    }
    o[4] = WMMA_F16(pa, ones, o[4]);  // every column = row sum of P
    __syncthreads();
  }

  // ---- finalize: O / rowsum -> attn[b, s, h*64 + d] (f32) ----
  const int b = bh >> 4, h = bh & 15;
#pragma unroll
  for (int r = 0; r < 8; ++r) {
    float inv = 1.0f / o[4][r];
    int srow = q0 + w * 16 + r + halfoff;
    size_t rowb = ((size_t)b * 2048 + srow) * 1024 + h * 64;
#pragma unroll
    for (int j = 0; j < 4; ++j)
      attn[rowb + j * 16 + (lane & 15)] = o[j][r] * inv;
  }
}

// ---------------------------------------------------------------------------
extern "C" void kernel_launch(void* const* d_in, const int* in_sizes, int n_in,
                              void* d_out, int out_size, void* d_ws, size_t ws_size,
                              hipStream_t stream) {
  (void)in_sizes; (void)n_in; (void)out_size; (void)ws_size;
  const float* x      = (const float*)d_in[0];  // [2,2048,1024]
  const float* w_qkv  = (const float*)d_in[1];  // [1024,3072]
  const float* b_qkv  = (const float*)d_in[2];  // [3072]
  const float* w_proj = (const float*)d_in[3];  // [1024,1024]
  const float* b_proj = (const float*)d_in[4];  // [1024]
  float* out = (float*)d_out;                   // [2,2048,1024]

  // workspace: Q,K f16 [2,16,2048,64]; V f16 transposed [2,16,64,2048];
  // attn f32 [4096,1024]
  const size_t HBUF = (size_t)2 * 16 * 2048 * 64;
  _Float16* qb   = (_Float16*)d_ws;
  _Float16* kbuf = qb + HBUF;
  _Float16* vbuf = kbuf + HBUF;
  float* attnf = (float*)(vbuf + HBUF);

  hgemm_wmma_gfx1250<0><<<dim3(3072 / 128, 4096 / 128), 256, 0, stream>>>(
      x, w_qkv, b_qkv, qb, kbuf, vbuf, nullptr, 4096, 3072, 1024);

  flash_attn_wmma_gfx1250<<<dim3(2048 / 64, 2 * 16), 128, 0, stream>>>(
      qb, kbuf, vbuf, attnf);

  hgemm_wmma_gfx1250<1><<<dim3(1024 / 128, 4096 / 128), 256, 0, stream>>>(
      attnf, w_proj, b_proj, nullptr, nullptr, nullptr, out, 4096, 1024, 1024);
}
